// WindowAttention_45277545234580
// MI455X (gfx1250) — matrix operand order repaired and checked by reading, not compiled, not run on hardware
//
#include <hip/hip_runtime.h>
#include <hip/hip_bf16.h>

#define DIM    384
#define TRIPLE 1152          // 3*DIM
#define NTOK   49
#define NPAD   64
#define NHEAD  12
#define HD     32
#define NWIN   64
#define MASKLD 52
#define QSCALE 0.17677669529663687f   // 1/sqrt(32)

typedef __bf16 bf16_t;
typedef __attribute__((ext_vector_type(8)))  __bf16 v8bf;
typedef __attribute__((ext_vector_type(16))) __bf16 v16bf;
typedef __attribute__((ext_vector_type(8)))  float  v8f;

// ---------------------------------------------------------------------------
// A-fragment (16x32 bf16, M x K). Contiguous k order used here (B matches):
//  lane m = lane%16 ; elems j=0..15 -> K = khalf + j
//  where khalf = (lane>=16) ? 16 : 0.  base points at (row0, k0), row-major, ld elems.
__device__ __forceinline__ v16bf load_frag_a(const bf16_t* base, int ld, int lane) {
    int m     = lane & 15;
    int khalf = (lane >> 4) << 4;
    const bf16_t* p = base + m * ld + khalf;
    v8bf lo = *(const v8bf*)(p);
    v8bf hi = *(const v8bf*)(p + 8);
    v16bf a;
#pragma unroll
    for (int j = 0; j < 8; ++j) { a[j] = lo[j]; a[8 + j] = hi[j]; }
    return a;
}

// B-fragment (32x16 bf16, K x N) where source is stored "N-major":
//  row n (= output column) is contiguous over K.
//  lane n = lane%16 ; elems j=0..15 -> K = ksel + j, ksel = (lane>=16) ? 16 : 0.
__device__ __forceinline__ v16bf load_frag_b(const bf16_t* base, int ld, int lane) {
    int n    = lane & 15;
    int ksel = (lane >> 4) << 4;
    const bf16_t* p = base + n * ld + ksel;
    v8bf lo = *(const v8bf*)(p);
    v8bf hi = *(const v8bf*)(p + 8);
    v16bf b;
#pragma unroll
    for (int j = 0; j < 8; ++j) { b[j] = lo[j]; b[8 + j] = hi[j]; }
    return b;
}

#define WMMA_BF16(A, B, C) \
    __builtin_amdgcn_wmma_f32_16x16x32_bf16(false, (A), false, (B), (short)0, (C), false, false)

// ---------------------------------------------------------------------------
__global__ void winattn_prep_weights(const float* __restrict__ qkv_w,
                                     const float* __restrict__ proj_w,
                                     bf16_t* __restrict__ wqkvT,
                                     bf16_t* __restrict__ wprojT) {
    int stride = gridDim.x * blockDim.x;
    int tid = blockIdx.x * blockDim.x + threadIdx.x;
    // qkv_w [DIM, 3DIM] -> wqkvT [3DIM, DIM] bf16
    for (int i = tid; i < DIM * TRIPLE; i += stride) {
        int o = i / DIM, k = i - o * DIM;
        wqkvT[i] = (bf16_t)qkv_w[k * TRIPLE + o];
    }
    // proj_w [DIM, DIM] -> wprojT [DIM, DIM] bf16 (transposed)
    for (int i = tid; i < DIM * DIM; i += stride) {
        int o = i / DIM, k = i - o * DIM;
        wprojT[i] = (bf16_t)proj_w[k * DIM + o];
    }
}

// ---------------------------------------------------------------------------
__global__ __launch_bounds__(256) void winattn_fused(
    const float* __restrict__ x,       // [B, 49, 384]
    const float* __restrict__ mask,    // [64, 49, 49]
    const float* __restrict__ qkv_b,   // [1152]
    const float* __restrict__ proj_b,  // [384]
    const bf16_t* __restrict__ wqkvT,  // [1152, 384] bf16
    const bf16_t* __restrict__ wprojT, // [384, 384] bf16
    float* __restrict__ out)           // [B, 49, 384]
{
    __shared__ bf16_t sXO[NPAD * DIM];        // 48 KB : X (phase 1), O (phase 3)
    __shared__ bf16_t sQKV[NPAD * TRIPLE];    // 144 KB
    __shared__ bf16_t sP[8 * 16 * 64];        // 16 KB : per-wave P strips
    __shared__ float  sMask[NTOK * MASKLD];   // ~10 KB

    const int b    = blockIdx.x;
    const int tid  = threadIdx.x;
    const int wave = tid >> 5;
    const int lane = tid & 31;
    const int w    = b & (NWIN - 1);
    const int cc   = lane & 15;           // column within tile (C layout)
    const int mb   = (lane >> 4) << 3;    // row-half offset (C layout)

    // ---- stage X (bf16, zero-padded to 64 rows) and mask ----
    for (int idx = tid; idx < NPAD * DIM; idx += 256) {
        int r = idx / DIM, c = idx - r * DIM;
        float v = (r < NTOK) ? x[(size_t)b * NTOK * DIM + r * DIM + c] : 0.0f;
        sXO[idx] = (bf16_t)v;
    }
    for (int idx = tid; idx < NTOK * NTOK; idx += 256) {
        int i = idx / NTOK, j = idx - i * NTOK;
        sMask[i * MASKLD + j] = mask[(size_t)w * NTOK * NTOK + idx];
    }
    __syncthreads();

    // ---- phase 1: QKV = X @ Wqkv + b ----
    // 4 M-tiles x 72 N-tiles; wave w owns mi = w/2, nj in [(w&1)*36, +36).
    // A-fragments for the whole K dim are loaded ONCE per wave and reused.
    {
        const int mi  = wave >> 1;
        const int nj0 = (wave & 1) * 36;
        v16bf afr[12];
#pragma unroll
        for (int kk = 0; kk < 12; ++kk)
            afr[kk] = load_frag_a(&sXO[(mi * 16) * DIM + kk * 32], DIM, lane);
        for (int nj = nj0; nj < nj0 + 36; ++nj) {
            v8f acc = {};
#pragma unroll
            for (int kk = 0; kk < 12; ++kk) {
                v16bf bb = load_frag_b(&wqkvT[(size_t)(nj * 16) * DIM + kk * 32], DIM, lane);
                acc = WMMA_BF16(afr[kk], bb, acc);
            }
            int col = nj * 16 + cc;
            float bias = qkv_b[col];
            float sc = (col < DIM) ? QSCALE : 1.0f;   // scale Q only
#pragma unroll
            for (int r = 0; r < 8; ++r)
                sQKV[(mi * 16 + mb + r) * TRIPLE + col] = (bf16_t)((acc[r] + bias) * sc);
        }
    }
    __syncthreads();

    // ---- phase 2: attention per head ----
    for (int h = wave; h < NHEAD; h += 8) {
        bf16_t* sPw = &sP[wave * 16 * 64];

        // Per-head invariants hoisted out of the M-strip loop:
        // K^T B-fragments (one per N tile; K dim = head dim = 32)
        v16bf kb[4];
#pragma unroll
        for (int nj = 0; nj < 4; ++nj)
            kb[nj] = load_frag_b(&sQKV[(nj * 16) * TRIPLE + DIM + h * HD], TRIPLE, lane);
        // V B-fragments (gathered; [kt][nd])
        v16bf vb[2][2];
        {
            int ksel = (lane >> 4) << 4;
#pragma unroll
            for (int kt = 0; kt < 2; ++kt)
#pragma unroll
                for (int nd = 0; nd < 2; ++nd) {
                    v16bf t;
#pragma unroll
                    for (int j = 0; j < 16; ++j)
                        t[j] = sQKV[(kt * 32 + ksel + j) * TRIPLE + 2 * DIM + h * HD + nd * 16 + cc];
                    vb[kt][nd] = t;
                }
        }

        for (int mi = 0; mi < 4; ++mi) {
            // S strip (16 x 64): one WMMA (K = 32) per N tile
            v16bf qa = load_frag_a(&sQKV[(mi * 16) * TRIPLE + h * HD], TRIPLE, lane);
            v8f S[4];
#pragma unroll
            for (int nj = 0; nj < 4; ++nj) {
                v8f z = {};
                S[nj] = WMMA_BF16(qa, kb[nj], z);
            }
            // softmax over each row (rows live across 16-lane groups in C layout)
#pragma unroll
            for (int r = 0; r < 8; ++r) {
                int i = mi * 16 + mb + r;
                float v[4];
#pragma unroll
                for (int nj = 0; nj < 4; ++nj) {
                    int j = nj * 16 + cc;
                    float val = S[nj][r];
                    if (j < NTOK) { if (i < NTOK) val += sMask[i * MASKLD + j]; }
                    else val = -1e30f;
                    v[nj] = val;
                }
                float mx = fmaxf(fmaxf(v[0], v[1]), fmaxf(v[2], v[3]));
#pragma unroll
                for (int off = 1; off < 16; off <<= 1)
                    mx = fmaxf(mx, __shfl_xor(mx, off, 32));
                float e[4], sum = 0.0f;
#pragma unroll
                for (int nj = 0; nj < 4; ++nj) {
                    e[nj] = (v[nj] > -1e29f) ? __expf(v[nj] - mx) : 0.0f;
                    sum += e[nj];
                }
#pragma unroll
                for (int off = 1; off < 16; off <<= 1)
                    sum += __shfl_xor(sum, off, 32);
                float inv = 1.0f / sum;
#pragma unroll
                for (int nj = 0; nj < 4; ++nj)
                    sPw[(mb + r) * 64 + nj * 16 + cc] = (bf16_t)(e[nj] * inv);
            }
            // O strip = P (16x64) @ V_h (64x32)
#pragma unroll
            for (int nd = 0; nd < 2; ++nd) {
                v8f o = {};
#pragma unroll
                for (int kt = 0; kt < 2; ++kt) {
                    v16bf pa = load_frag_a(&sPw[kt * 32], 64, lane);
                    o = WMMA_BF16(pa, vb[kt][nd], o);
                }
#pragma unroll
                for (int r = 0; r < 8; ++r)
                    sXO[(mi * 16 + mb + r) * DIM + h * HD + nd * 16 + cc] = (bf16_t)o[r];
            }
        }
    }
    __syncthreads();

    // ---- phase 3: out = O @ Wproj + b ----
    // 4 M-tiles x 24 N-tiles; wave w owns mi = w/2, nj in [(w&1)*12, +12).
    {
        const int mi  = wave >> 1;
        const int nj0 = (wave & 1) * 12;
        v16bf afr[12];
#pragma unroll
        for (int kk = 0; kk < 12; ++kk)
            afr[kk] = load_frag_a(&sXO[(mi * 16) * DIM + kk * 32], DIM, lane);
        for (int nj = nj0; nj < nj0 + 12; ++nj) {
            v8f acc = {};
#pragma unroll
            for (int kk = 0; kk < 12; ++kk) {
                v16bf bb = load_frag_b(&wprojT[(size_t)(nj * 16) * DIM + kk * 32], DIM, lane);
                acc = WMMA_BF16(afr[kk], bb, acc);
            }
            int col = nj * 16 + cc;
            float bias = proj_b[col];
#pragma unroll
            for (int r = 0; r < 8; ++r) {
                int m = mi * 16 + mb + r;
                if (m < NTOK)
                    out[(size_t)b * NTOK * DIM + m * DIM + col] = acc[r] + bias;
            }
        }
    }
}

// ---------------------------------------------------------------------------
extern "C" void kernel_launch(void* const* d_in, const int* in_sizes, int n_in,
                              void* d_out, int out_size, void* d_ws, size_t ws_size,
                              hipStream_t stream) {
    const float* x      = (const float*)d_in[0];
    const float* mask   = (const float*)d_in[1];
    // d_in[2] = num_heads (scalar, fixed at 12 here)
    const float* qkv_w  = (const float*)d_in[3];
    const float* qkv_b  = (const float*)d_in[4];
    const float* proj_w = (const float*)d_in[5];
    const float* proj_b = (const float*)d_in[6];

    bf16_t* wqkvT  = (bf16_t*)d_ws;                 // 1152*384 bf16
    bf16_t* wprojT = wqkvT + (size_t)TRIPLE * DIM;  // 384*384 bf16

    winattn_prep_weights<<<256, 256, 0, stream>>>(qkv_w, proj_w, wqkvT, wprojT);

    winattn_fused<<<4096, 256, 0, stream>>>(x, mask, qkv_b, proj_b,
                                            wqkvT, wprojT, (float*)d_out);
}
